// KA_GCN_latent_6863357739560
// MI455X (gfx1250) — matrix operand order repaired and checked by reading, not compiled, not run on hardware
//
#include <hip/hip_runtime.h>

typedef _Float16 v16h __attribute__((ext_vector_type(16)));
typedef _Float16 v8h  __attribute__((ext_vector_type(8)));
typedef float    v8f  __attribute__((ext_vector_type(8)));

#define HARM 8
#define HID_DIM 128

// ---------------------------------------------------------------- utilities
__global__ void fill_f32(float* p, float v, long long n) {
  long long i = (long long)blockIdx.x * blockDim.x + threadIdx.x;
  if (i < n) p[i] = v;
}

// Pack W [2, out, in, H] (f32) into WMMA-B-fragment-ordered f16:
// idx = ((kc*notile + otile)*32 + lane)*16 + j
// lane holds col = otile*16 + lane%16, K = kc*32 + (lane/16)*8 + (j%8) + (j/8)*16
__global__ void pack_weights(const float* __restrict__ W, _Float16* __restrict__ Bp,
                             int indim, int out, long long total) {
  long long idx = (long long)blockIdx.x * blockDim.x + threadIdx.x;
  if (idx >= total) return;
  int notile = out >> 4;
  long long per_kc = (long long)notile * 512;
  int kc    = (int)(idx / per_kc);
  int r     = (int)(idx - (long long)kc * per_kc);
  int otile = r >> 9;
  int r2    = r & 511;
  int lane  = r2 >> 4;
  int j     = r2 & 15;
  int K = kc * 32 + ((lane >> 4) << 3) + (j & 7) + ((j >> 3) << 4);       // k index 0..2*in*H-1
  int o = (otile << 4) + (lane & 15);
  int ih  = indim * HARM;
  int s   = (K >= ih) ? 1 : 0;                    // 0 = cos block, 1 = sin block
  int rem = K - s * ih;
  int i   = rem >> 3;                             // input feature
  int t   = rem & 7;                              // harmonic-1
  Bp[idx] = (_Float16)W[(((long long)s * out + o) * indim + i) * HARM + t];
}

// ---------------------------------------------------------------- KAN GEMM
// Y[M, out] = cos/sin-basis(X[M, indim]) @ Bp ; K = 2*indim*HARM (mult of 512)
// block: 256 thr = 8 waves; block owns 16 rows; wave w owns out cols [16w,16w+16)
__global__ __launch_bounds__(256)
void kan_wmma(const float* __restrict__ X, const _Float16* __restrict__ Bp,
              float* __restrict__ Y, int indim, int out, int M) {
  __shared__ _Float16 Alds[16 * 512];   // 16 rows x 512 K-chunk of basis (16 KB)
  __shared__ float    Xs[16 * HID_DIM]; // input rows (8 KB)

  const int tid  = threadIdx.x;
  const int lane = tid & 31;
  const int wave = tid >> 5;
  const int tile = blockIdx.x;
  const int K      = 2 * indim * HARM;
  const int nchunk = K >> 9;
  const int notile = out >> 4;
  const int ih     = indim * HARM;
  const bool active = (wave < notile);

  // stage X rows (clamp for safe reads on a ragged last tile)
  for (int idx = tid; idx < 16 * indim; idx += 256) {
    int row = idx / indim, coli = idx - row * indim;
    int g = tile * 16 + row; if (g >= M) g = M - 1;
    Xs[row * indim + coli] = X[(long long)g * indim + coli];
  }
  __syncthreads();

  v8f acc;
#pragma unroll
  for (int j = 0; j < 8; ++j) acc[j] = 0.0f;

  for (int c = 0; c < nchunk; ++c) {
    // fill basis chunk: A[row][kk] = cos/sin(x[row,i] * (t+1))
    for (int idx = tid; idx < 16 * 512; idx += 256) {
      int row = idx >> 9, kk = idx & 511;
      int kg  = (c << 9) + kk;
      int s   = (kg >= ih) ? 1 : 0;
      int rem = kg - s * ih;
      int i   = rem >> 3, t = rem & 7;
      float ang = Xs[row * indim + i] * (float)(t + 1);
      Alds[(row << 9) + kk] = (_Float16)(s ? __sinf(ang) : __cosf(ang));
    }
    __syncthreads();

    if (active) {
      // A frag per ISA: lane<16 -> K-offs {0..7,16..23}; lane>=16 -> {8..15,24..31}
      const _Float16* arow = Alds + ((lane & 15) << 9) + ((lane >> 4) << 3);
#pragma unroll
      for (int ks = 0; ks < 16; ++ks) {
        v8h a0 = *(const v8h*)(arow + ks * 32);        // ds_load_b128
        v8h a1 = *(const v8h*)(arow + ks * 32 + 16);   // ds_load_b128
        v16h a;
#pragma unroll
        for (int j = 0; j < 8; ++j) { a[j] = a0[j]; a[j + 8] = a1[j]; }

        long long kc = (long long)c * 16 + ks;
        const _Float16* bptr = Bp + ((kc * notile + wave) * 32 + lane) * 16;
        __builtin_prefetch(bptr + (long long)notile * 512, 0, 1); // next K-step
        v16h b = *(const v16h*)bptr;                   // 32B global load

        acc = __builtin_amdgcn_wmma_f32_16x16x32_f16(
            false, a, false, b, (short)0, acc, false, false);
      }
    }
    __syncthreads();
  }

  if (active) {
    int col = (wave << 4) + (lane & 15);
    int rb  = (lane < 16) ? 0 : 8;                     // C/D layout: M = v + rb
#pragma unroll
    for (int v = 0; v < 8; ++v) {
      int g = tile * 16 + rb + v;
      if (g < M) Y[(long long)g * out + col] = acc[v];
    }
  }
}

// ---------------------------------------------------------------- graph ops
__global__ void deg_edges(const int* __restrict__ col, float* deg, int E) {
  int e = blockIdx.x * blockDim.x + threadIdx.x;
  if (e < E) atomicAdd(&deg[col[e]], 1.0f);
}

__global__ void make_dinv(float* d, int n) {
  int i = blockIdx.x * blockDim.x + threadIdx.x;
  if (i < n) d[i] = rsqrtf(d[i]);   // deg >= 1 thanks to self-loop init
}

// one wave per edge (E real edges then N self-loops); lane carries one float4
__global__ void aggregate(const float* __restrict__ T, const int* __restrict__ row,
                          const int* __restrict__ colx, const float* __restrict__ dinv,
                          float* __restrict__ Hout, int E, int N) {
  long long gt = (long long)blockIdx.x * blockDim.x + threadIdx.x;
  int w = (int)(gt >> 5), lane = threadIdx.x & 31;
  if (w >= E + N) return;
  int r, c;
  if (w < E) { r = row[w]; c = colx[w]; } else { r = c = w - E; }
  float nm = dinv[r] * dinv[c];
  float4 m = ((const float4*)(T + (long long)r * HID_DIM))[lane];
  float* dst = Hout + (long long)c * HID_DIM + lane * 4;
  atomicAdd(dst + 0, nm * m.x);
  atomicAdd(dst + 1, nm * m.y);
  atomicAdd(dst + 2, nm * m.z);
  atomicAdd(dst + 3, nm * m.w);
}

__global__ void pool_accum(const float* __restrict__ Hf, const int* __restrict__ batch,
                           float* sums, float* cnt, int N) {
  long long gt = (long long)blockIdx.x * blockDim.x + threadIdx.x;
  int w = (int)(gt >> 5), lane = threadIdx.x & 31;
  if (w >= N) return;
  int g = batch[w];
  float4 v = ((const float4*)(Hf + (long long)w * HID_DIM))[lane];
  float* dst = sums + (long long)g * HID_DIM + lane * 4;
  atomicAdd(dst + 0, v.x);
  atomicAdd(dst + 1, v.y);
  atomicAdd(dst + 2, v.z);
  atomicAdd(dst + 3, v.w);
  if (lane == 0) atomicAdd(&cnt[g], 1.0f);
}

__global__ void pool_final(const float* sums, const float* cnt, float* y, int total) {
  int i = blockIdx.x * blockDim.x + threadIdx.x;
  if (i < total) y[i] = sums[i] / fmaxf(cnt[i >> 7], 1.0f);
}

// ---------------------------------------------------------------- launcher
extern "C" void kernel_launch(void* const* d_in, const int* in_sizes, int n_in,
                              void* d_out, int out_size, void* d_ws, size_t ws_size,
                              hipStream_t stream) {
  const float* features = (const float*)d_in[0];
  const int*   ei       = (const int*)d_in[1];   // [2,E] row-major
  const int*   batch    = (const int*)d_in[2];
  const float* Wemb     = (const float*)d_in[3]; // [2,128,64,8]
  const float* Wmp      = (const float*)d_in[4]; // [3,2,128,128,8]
  const float* Wread    = (const float*)d_in[5]; // [2,32,128,8]

  const int IN = 64, OUT = HID_DIM, LAT = 32, G = 64, L = 3;
  const int N = in_sizes[0] / IN;
  const int E = in_sizes[1] / 2;
  const int* erow = ei;
  const int* ecol = ei + E;

  char* ws = (char*)d_ws;
  size_t off = 0;
  auto carve = [&](size_t bytes) { char* p = ws + off; off += (bytes + 255) & ~(size_t)255; return p; };
  _Float16* BpE = (_Float16*)carve((size_t)1024 * 128 * 2);
  _Float16* Bp0 = (_Float16*)carve((size_t)2048 * 128 * 2);
  _Float16* Bp1 = (_Float16*)carve((size_t)2048 * 128 * 2);
  _Float16* Bp2 = (_Float16*)carve((size_t)2048 * 128 * 2);
  _Float16* BpR = (_Float16*)carve((size_t)2048 * 32 * 2);
  float* deg  = (float*)carve((size_t)N * 4);
  float* hA   = (float*)carve((size_t)N * HID_DIM * 4);
  float* hB   = (float*)carve((size_t)N * HID_DIM * 4);
  float* sums = (float*)carve((size_t)G * HID_DIM * 4);
  float* cnt  = (float*)carve((size_t)G * 4);
  float* ypool= (float*)carve((size_t)G * HID_DIM * 4);
  _Float16* Bps[3] = {Bp0, Bp1, Bp2};

  // 1) pack all weights into WMMA B-fragment layout (f16)
  {
    long long tE = (long long)1024 * 128;
    pack_weights<<<(int)((tE + 255) / 256), 256, 0, stream>>>(Wemb, BpE, IN, OUT, tE);
    long long tM = (long long)2048 * 128;
    for (int l = 0; l < L; ++l)
      pack_weights<<<(int)((tM + 255) / 256), 256, 0, stream>>>(
          Wmp + (long long)l * 2 * HID_DIM * HID_DIM * HARM, Bps[l], HID_DIM, OUT, tM);
    long long tR = (long long)2048 * 32;
    pack_weights<<<(int)((tR + 255) / 256), 256, 0, stream>>>(Wread, BpR, HID_DIM, LAT, tR);
  }

  // 2) degrees (init 1.0 for self-loop) -> dinv = deg^-1/2
  fill_f32<<<(N + 255) / 256, 256, 0, stream>>>(deg, 1.0f, N);
  deg_edges<<<(E + 255) / 256, 256, 0, stream>>>(ecol, deg, E);
  make_dinv<<<(N + 255) / 256, 256, 0, stream>>>(deg, N);

  const int mtiles = (N + 15) / 16;

  // 3) node embedding: hA = kan(features, W_embed)
  kan_wmma<<<mtiles, 256, 0, stream>>>(features, BpE, hA, IN, OUT, N);

  // 4) message-passing layers: hB = kan(hA); hA = scatter-add(norm * hB[row])
  for (int l = 0; l < L; ++l) {
    kan_wmma<<<mtiles, 256, 0, stream>>>(hA, Bps[l], hB, HID_DIM, OUT, N);
    fill_f32<<<(int)(((long long)N * HID_DIM + 255) / 256), 256, 0, stream>>>(
        hA, 0.0f, (long long)N * HID_DIM);
    int waves = E + N;
    aggregate<<<(waves + 7) / 8, 256, 0, stream>>>(hB, erow, ecol, deg, hA, E, N);
  }

  // 5) global mean pool
  fill_f32<<<(G * HID_DIM + 255) / 256, 256, 0, stream>>>(sums, 0.0f, G * HID_DIM);
  fill_f32<<<1, 256, 0, stream>>>(cnt, 0.0f, G);
  pool_accum<<<(N + 7) / 8, 256, 0, stream>>>(hA, batch, sums, cnt, N);
  pool_final<<<(G * HID_DIM + 255) / 256, 256, 0, stream>>>(sums, cnt, ypool, G * HID_DIM);

  // 6) readout: d_out = kan(ypool, W_read)  [64 x 32] f32
  kan_wmma<<<G / 16, 256, 0, stream>>>(ypool, BpR, (float*)d_out, HID_DIM, LAT, G);
}
